// DIFFormer_82274393522377
// MI455X (gfx1250) — matrix operand order repaired and checked by reading, not compiled, not run on hardware
//
#include <hip/hip_runtime.h>
#include <hip/hip_bf16.h>

typedef __bf16 bf16;
typedef __attribute__((ext_vector_type(16))) __bf16 v16bf;
typedef __attribute__((ext_vector_type(8)))  __bf16 v8bf;
typedef __attribute__((ext_vector_type(8)))  float  v8f;
typedef __attribute__((__vector_size__(16))) int    v4i;

#define AS1 __attribute__((address_space(1)))
#define AS3 __attribute__((address_space(3)))

#if defined(__AMDGCN__) && __has_builtin(__builtin_amdgcn_global_load_async_to_lds_b128)
#define HAVE_ASYNC_LDS 1
#else
#define HAVE_ASYNC_LDS 0
#endif

#define HCDIM 256
#define NHEAD 4
#define D2DIM 512
#define QKVD  1024            // NHEAD*HCDIM
#define QKV3  3072            // packed q|k|v width
#define OUTD  64
#define EPSV  1e-5f
#define KCH   8192            // split-K chunk (multiple of 32)

// ---------------- WMMA GEMM:  C[M][N] = A[M][K] * BT[N][K]^T (+bias) ----------
#define BM 64
#define BN 128
#define BK 32
#define LDSTR 40              // row stride in elements (80B: 16B-aligned, conflict-free)

__device__ __forceinline__ v8bf zero8() {
  v8bf z;
#pragma unroll
  for (int i = 0; i < 8; ++i) z[i] = (bf16)0.f;
  return z;
}

// async copy of one 32B group (two b128) straight into LDS, no VGPR staging
__device__ __forceinline__ void cp32_async(const bf16* g, bf16* l) {
#if HAVE_ASYNC_LDS
  AS1 v4i* gp = (AS1 v4i*)(bf16*)g;
  AS3 v4i* lp = (AS3 v4i*)l;
  __builtin_amdgcn_global_load_async_to_lds_b128(gp, lp, 0, 0);
  __builtin_amdgcn_global_load_async_to_lds_b128(gp, lp, 16, 0);
#else
  v8bf* lp = (v8bf*)l;
  const v8bf* gp = (const v8bf*)g;
  lp[0] = gp[0];
  lp[1] = gp[1];
#endif
}
__device__ __forceinline__ void zero32_lds(bf16* l) {
  v8bf* lp = (v8bf*)l;
  lp[0] = zero8();
  lp[1] = zero8();
}
__device__ __forceinline__ void wait_async() {
#if HAVE_ASYNC_LDS
#if __has_builtin(__builtin_amdgcn_s_wait_asynccnt)
  __builtin_amdgcn_s_wait_asynccnt(0);
#else
  asm volatile("s_wait_asynccnt 0x0" ::: "memory");
#endif
#endif
}

// A fragment: lane = M row (lane&15); K runs [kb,kb+7],[kb+16,kb+23], kb=0/8
__device__ __forceinline__ v16bf frag_a(const bf16* As, int m0, int lane) {
  int m  = m0 + (lane & 15);
  int kb = (lane < 16) ? 0 : 8;
  v8bf lo = *(const v8bf*)(As + m * LDSTR + kb);
  v8bf hi = *(const v8bf*)(As + m * LDSTR + kb + 16);
  return __builtin_shufflevector(lo, hi, 0, 1, 2, 3, 4, 5, 6, 7, 8, 9, 10, 11, 12, 13, 14, 15);
}
// B fragment: lane = N col (lane&15); K runs [kb,kb+7],[kb+16,kb+23], kb=0/8
__device__ __forceinline__ v16bf frag_b(const bf16* Bs, int n0, int lane) {
  int n  = n0 + (lane & 15);
  int kb = (lane < 16) ? 0 : 8;
  v8bf lo = *(const v8bf*)(Bs + n * LDSTR + kb);
  v8bf hi = *(const v8bf*)(Bs + n * LDSTR + kb + 16);
  return __builtin_shufflevector(lo, hi, 0, 1, 2, 3, 4, 5, 6, 7, 8, 9, 10, 11, 12, 13, 14, 15);
}

// Requirements: lda/ldb multiples of 8 elements (16B rows), K-chunk boundaries
// multiples of 16 (512/256/50000/8192 all qualify).
template <bool ATOMIC_ACC>
__global__ __launch_bounds__(128) void gemm_wmma_bf16(
    const bf16* __restrict__ A, const bf16* __restrict__ BT,
    const float* __restrict__ bias, float* __restrict__ C,
    int M, int N, int K, int lda, int ldb, int ldc, int kchunk) {
  __shared__ __align__(16) bf16 As[2][BM * LDSTR];
  __shared__ __align__(16) bf16 Bs[2][BN * LDSTR];

  const int tid  = threadIdx.x;
  const int lane = tid & 31;
  const int wid  = tid >> 5;
  const int mt   = blockIdx.x * BM;
  const int nt   = blockIdx.y * BN;
  const int k0   = blockIdx.z * kchunk;
  int k1 = k0 + kchunk;
  if (k1 > K) k1 = K;

  const int mw = (wid >> 1) * 32;   // wave tile: 32 rows x 64 cols
  const int nw = (wid & 1) * 64;

  // stage one BK-deep tile pair into buffer `buf` (async when in-bounds)
  auto stage = [&](int buf, int kt) {
    {  // A: 64 rows x 32 K, one 32B group per thread
      int ml = tid >> 1, half = (tid & 1) * 16;
      int mg = mt + ml, kg = kt + half;
      bf16* lp = As[buf] + ml * LDSTR + half;
      if (mg < M && kg < k1)
        cp32_async(A + (size_t)mg * lda + kg, lp);
      else
        zero32_lds(lp);
    }
#pragma unroll
    for (int g = 0; g < 2; ++g) {  // B: 128 rows x 32 K, two 32B groups per thread
      int idx = tid + g * 128;
      int nl = idx >> 1, half = (idx & 1) * 16;
      int ng = nt + nl, kg = kt + half;
      bf16* lp = Bs[buf] + nl * LDSTR + half;
      if (ng < N && kg < k1)
        cp32_async(BT + (size_t)ng * ldb + kg, lp);
      else
        zero32_lds(lp);
    }
  };

  const v8f zf = {0.f, 0.f, 0.f, 0.f, 0.f, 0.f, 0.f, 0.f};
  v8f acc[2][4] = {{zf, zf, zf, zf}, {zf, zf, zf, zf}};

  int buf = 0;
  stage(0, k0);
  wait_async();
  __syncthreads();

  for (int kt = k0; kt < k1; kt += BK) {
    if (kt + BK < k1) stage(buf ^ 1, kt + BK);  // prefetch next tile while computing

    const bf16* Ab = As[buf];
    const bf16* Bb = Bs[buf];
    v16bf a0 = frag_a(Ab, mw + 0, lane);
    v16bf a1 = frag_a(Ab, mw + 16, lane);
    v16bf b0 = frag_b(Bb, nw + 0, lane);
    v16bf b1 = frag_b(Bb, nw + 16, lane);
    v16bf b2 = frag_b(Bb, nw + 32, lane);
    v16bf b3 = frag_b(Bb, nw + 48, lane);
    acc[0][0] = __builtin_amdgcn_wmma_f32_16x16x32_bf16(false, a0, false, b0, (short)0, acc[0][0], false, false);
    acc[0][1] = __builtin_amdgcn_wmma_f32_16x16x32_bf16(false, a0, false, b1, (short)0, acc[0][1], false, false);
    acc[0][2] = __builtin_amdgcn_wmma_f32_16x16x32_bf16(false, a0, false, b2, (short)0, acc[0][2], false, false);
    acc[0][3] = __builtin_amdgcn_wmma_f32_16x16x32_bf16(false, a0, false, b3, (short)0, acc[0][3], false, false);
    acc[1][0] = __builtin_amdgcn_wmma_f32_16x16x32_bf16(false, a1, false, b0, (short)0, acc[1][0], false, false);
    acc[1][1] = __builtin_amdgcn_wmma_f32_16x16x32_bf16(false, a1, false, b1, (short)0, acc[1][1], false, false);
    acc[1][2] = __builtin_amdgcn_wmma_f32_16x16x32_bf16(false, a1, false, b2, (short)0, acc[1][2], false, false);
    acc[1][3] = __builtin_amdgcn_wmma_f32_16x16x32_bf16(false, a1, false, b3, (short)0, acc[1][3], false, false);

    wait_async();     // staged tile fully in LDS
    __syncthreads();  // everyone done reading `buf`, staged buffer published
    buf ^= 1;
  }

  // C/D: lane = N col, VGPR i -> M = i (lanes<16) / i+8 (lanes>=16)
#pragma unroll
  for (int ti = 0; ti < 2; ++ti) {
#pragma unroll
    for (int tj = 0; tj < 4; ++tj) {
      int ncol = nt + nw + tj * 16 + (lane & 15);
      if (ncol >= N) continue;
      float badd = (!ATOMIC_ACC && bias) ? bias[ncol] : 0.f;
#pragma unroll
      for (int i = 0; i < 8; ++i) {
        int mrow = mt + mw + ti * 16 + ((lane < 16) ? i : i + 8);
        if (mrow < M) {
          float val = acc[ti][tj][i] + badd;
          if (ATOMIC_ACC)
            atomicAdd(&C[(size_t)mrow * ldc + ncol], val);
          else
            C[(size_t)mrow * ldc + ncol] = val;
        }
      }
    }
  }
}

// ---------------- elementwise / reduction kernels ----------------
__global__ void k_cast_bf16(const float* __restrict__ in, bf16* __restrict__ out, long n) {
  long i = (long)blockIdx.x * blockDim.x + threadIdx.x;
  if (i < n) out[i] = (bf16)in[i];
}
// in[K][N] (row-major) -> out[N][K]
__global__ void k_cast_bf16_t(const float* __restrict__ in, bf16* __restrict__ out,
                              int K, int N) {
  long i = (long)blockIdx.x * blockDim.x + threadIdx.x;
  if (i >= (long)K * N) return;
  int n = (int)(i / K), kk = (int)(i % K);
  out[(size_t)n * K + kk] = (bf16)in[(size_t)kk * N + n];
}
__global__ void k_pack3(const float* __restrict__ a, const float* __restrict__ b,
                        const float* __restrict__ c, float* __restrict__ o, int n) {
  int i = blockIdx.x * blockDim.x + threadIdx.x;
  if (i < n) o[i] = a[i];
  else if (i < 2 * n) o[i] = b[i - n];
  else if (i < 3 * n) o[i] = c[i - 2 * n];
}

__global__ void k_bn_relu(float* __restrict__ h, bf16* __restrict__ hb,
                          const float* g, const float* b, const float* mu,
                          const float* var, long n) {
  long i = (long)blockIdx.x * blockDim.x + threadIdx.x;
  if (i >= n) return;
  int c = (int)(i % D2DIM);
  float val = (h[i] - mu[c]) * rsqrtf(var[c] + EPSV) * g[c] + b[c];
  val = fmaxf(val, 0.f);
  h[i] = val;
  hb[i] = (bf16)val;
}

// one wave per (node, head): L2-normalize q,k from packed qkv[N][3072];
// q row-major bf16, k/v feature-major bf16 ([QKVD][Nn]); accumulate ksum/vsum.
__global__ void k_norm_qkv(const float* __restrict__ qkv, bf16* __restrict__ qnb,
                           bf16* __restrict__ knT, bf16* __restrict__ vT,
                           float* __restrict__ ksum, float* __restrict__ vsum, int Nn) {
  int wave = (int)((blockIdx.x * (long)blockDim.x + threadIdx.x) >> 5);
  int lane = threadIdx.x & 31;
  if (wave >= Nn * NHEAD) return;
  int n = wave / NHEAD, hh = wave % NHEAD;
  size_t inb = (size_t)n * QKV3 + hh * HCDIM;
  float qv[8], kv[8], vv[8], sq = 0.f, sk = 0.f;
#pragma unroll
  for (int i = 0; i < 8; ++i) {
    int c = lane + 32 * i;
    qv[i] = qkv[inb + c];
    kv[i] = qkv[inb + QKVD + c];
    vv[i] = qkv[inb + 2 * QKVD + c];
    sq += qv[i] * qv[i];
    sk += kv[i] * kv[i];
  }
#pragma unroll
  for (int off = 16; off > 0; off >>= 1) {
    sq += __shfl_xor(sq, off, 32);
    sk += __shfl_xor(sk, off, 32);
  }
  float rq = rsqrtf(sq), rk = rsqrtf(sk);
#pragma unroll
  for (int i = 0; i < 8; ++i) {
    int c = lane + 32 * i;
    int f = hh * HCDIM + c;
    float qn = qv[i] * rq, kn = kv[i] * rk;
    qnb[(size_t)n * QKVD + f] = (bf16)qn;
    knT[(size_t)f * Nn + n]   = (bf16)kn;
    vT[(size_t)f * Nn + n]    = (bf16)vv[i];
    atomicAdd(&ksum[f], kn);
    atomicAdd(&vsum[f], vv[i]);
  }
}

__global__ void k_den(const bf16* __restrict__ qnb, const float* __restrict__ ksum,
                      float* __restrict__ den, int Nn) {
  int wave = (int)((blockIdx.x * (long)blockDim.x + threadIdx.x) >> 5);
  int lane = threadIdx.x & 31;
  if (wave >= Nn * NHEAD) return;
  int n = wave / NHEAD, hh = wave % NHEAD;
  size_t base = (size_t)n * QKVD + hh * HCDIM;
  float s = 0.f;
#pragma unroll
  for (int i = 0; i < 8; ++i) {
    int c = lane + 32 * i;
    s += (float)qnb[base + c] * ksum[hh * HCDIM + c];
  }
#pragma unroll
  for (int off = 16; off > 0; off >>= 1) s += __shfl_xor(s, off, 32);
  if (lane == 0) den[n * NHEAD + hh] = s + (float)Nn;
}

__global__ void k_combine(const float* __restrict__ num, const float* __restrict__ vsum,
                          const float* __restrict__ den, float* __restrict__ x1, int Nn) {
  long i = (long)blockIdx.x * blockDim.x + threadIdx.x;
  if (i >= (long)Nn * HCDIM) return;
  int n = (int)(i / HCDIM), c = (int)(i % HCDIM);
  float s = 0.f;
#pragma unroll
  for (int hh = 0; hh < NHEAD; ++hh)
    s += (num[(size_t)n * QKVD + hh * HCDIM + c] + vsum[hh * HCDIM + c]) /
         den[n * NHEAD + hh];
  x1[i] = s * (1.f / NHEAD);
}

__global__ void k_deg_init(float* deg, int Nn) {
  int i = blockIdx.x * blockDim.x + threadIdx.x;
  if (i < Nn) deg[i] = 1.f;
}
__global__ void k_deg_acc(const int* __restrict__ dst, float* __restrict__ deg, int E) {
  int e = blockIdx.x * blockDim.x + threadIdx.x;
  if (e < E) atomicAdd(&deg[dst[e]], 1.f);
}
__global__ void k_dinv(const float* deg, float* dinv, int Nn) {
  int i = blockIdx.x * blockDim.x + threadIdx.x;
  if (i < Nn) dinv[i] = rsqrtf(fmaxf(deg[i], 1.f));
}
__global__ void k_gcn_self(const float* __restrict__ xw, const float* __restrict__ dinv,
                           float* __restrict__ gout, int Nn) {
  long i = (long)blockIdx.x * blockDim.x + threadIdx.x;
  if (i >= (long)Nn * HCDIM) return;
  int n = (int)(i / HCDIM);
  gout[i] = dinv[n] * dinv[n] * xw[i];
}
__global__ void k_gcn_scatter(const int* __restrict__ src, const int* __restrict__ dst,
                              const float* __restrict__ xw, const float* __restrict__ dinv,
                              float* __restrict__ gout, int E) {
  long t = (long)blockIdx.x * blockDim.x + threadIdx.x;
  if (t >= (long)E * 32) return;
  int e  = (int)(t >> 5);
  int c0 = (int)(t & 31) * 8;
  int s = src[e], d = dst[e];
  float nrm = dinv[s] * dinv[d];
  size_t sb = (size_t)s * HCDIM + c0, db = (size_t)d * HCDIM + c0;
#pragma unroll
  for (int j = 0; j < 8; ++j) atomicAdd(&gout[db + j], xw[sb + j] * nrm);
}

__global__ void k_concat_bn_relu(float* __restrict__ h, bf16* __restrict__ hb,
                                 const float* __restrict__ x1, const float* __restrict__ gout,
                                 const float* __restrict__ gcn_b, const float* g,
                                 const float* b, const float* mu, const float* var, long n) {
  long i = (long)blockIdx.x * blockDim.x + threadIdx.x;
  if (i >= n) return;
  int node = (int)(i / D2DIM), c = (int)(i % D2DIM);
  float val = (c < HCDIM) ? x1[(size_t)node * HCDIM + c]
                          : (gout[(size_t)node * HCDIM + (c - HCDIM)] + gcn_b[c - HCDIM]);
  float hn = 0.5f * val + 0.5f * h[i];
  hn = (hn - mu[c]) * rsqrtf(var[c] + EPSV) * g[c] + b[c];
  hn = fmaxf(hn, 0.f);
  h[i] = hn;
  hb[i] = (bf16)hn;
}

// ---------------- host orchestration ----------------
static inline dim3 cgrid(long n, int bs) { return dim3((unsigned)((n + bs - 1) / bs)); }

extern "C" void kernel_launch(void* const* d_in, const int* in_sizes, int n_in,
                              void* d_out, int out_size, void* d_ws, size_t ws_size,
                              hipStream_t stream) {
  const float* x     = (const float*)d_in[0];
  const int*   edges = (const int*)d_in[1];
  const int    Nn    = in_sizes[2];
  const int    E     = in_sizes[1] / 2;
  const float* fc0_w = (const float*)d_in[3];
  const float* fc0_b = (const float*)d_in[4];
  const float* wq    = (const float*)d_in[5];
  const float* bq    = (const float*)d_in[6];
  const float* wk    = (const float*)d_in[7];
  const float* bk    = (const float*)d_in[8];
  const float* wv    = (const float*)d_in[9];
  const float* bv    = (const float*)d_in[10];
  const float* gcn_w = (const float*)d_in[11];
  const float* gcn_b = (const float*)d_in[12];
  const float* bng   = (const float*)d_in[13];
  const float* bnb   = (const float*)d_in[14];
  const float* bnm   = (const float*)d_in[15];
  const float* bnv   = (const float*)d_in[16];
  const float* fow   = (const float*)d_in[17];
  const float* fobv  = (const float*)d_in[18];
  float* out = (float*)d_out;

  char* p = (char*)d_ws;
  auto alloc = [&](size_t bytes) -> void* {
    void* r = (void*)p;
    p += (bytes + 255) & ~(size_t)255;
    return r;
  };
  float* h    = (float*)alloc((size_t)Nn * D2DIM * 4);
  bf16*  hb   = (bf16*)alloc((size_t)Nn * D2DIM * 2);
  float* qkv  = (float*)alloc((size_t)Nn * QKV3 * 4);
  bf16*  qnb  = (bf16*)alloc((size_t)Nn * QKVD * 2);
  bf16*  knT  = (bf16*)alloc((size_t)QKVD * Nn * 2);
  bf16*  vT   = (bf16*)alloc((size_t)QKVD * Nn * 2);
  float* num  = (float*)alloc((size_t)Nn * QKVD * 4);
  float* den  = (float*)alloc((size_t)Nn * NHEAD * 4);
  float* x1   = (float*)alloc((size_t)Nn * HCDIM * 4);
  float* xw   = (float*)alloc((size_t)Nn * HCDIM * 4);
  float* gout = (float*)alloc((size_t)Nn * HCDIM * 4);
  float* deg  = (float*)alloc((size_t)Nn * 4);
  float* dinv = (float*)alloc((size_t)Nn * 4);
  float* PT   = (float*)alloc((size_t)NHEAD * HCDIM * HCDIM * 4);  // kvs^T per head
  bf16*  PTb  = (bf16*)alloc((size_t)NHEAD * HCDIM * HCDIM * 2);
  float* ksum = (float*)alloc((size_t)QKVD * 4);
  float* vsum = (float*)alloc((size_t)QKVD * 4);
  bf16* w0bT   = (bf16*)alloc((size_t)D2DIM * D2DIM * 2);
  bf16* wqkvT  = (bf16*)alloc((size_t)QKV3 * D2DIM * 2);
  bf16* gwbT   = (bf16*)alloc((size_t)HCDIM * D2DIM * 2);
  bf16* fowbT  = (bf16*)alloc((size_t)OUTD * D2DIM * 2);
  float* biasp = (float*)alloc((size_t)QKV3 * 4);

  // ---- stage 0: h = relu(bn0(x @ fc0_w + fc0_b)) ----
  k_cast_bf16<<<cgrid((long)Nn * D2DIM, 256), 256, 0, stream>>>(x, hb, (long)Nn * D2DIM);
  k_cast_bf16_t<<<cgrid((long)D2DIM * D2DIM, 256), 256, 0, stream>>>(fc0_w, w0bT, D2DIM, D2DIM);
  {
    dim3 g((Nn + BM - 1) / BM, D2DIM / BN, 1);
    gemm_wmma_bf16<false><<<g, 128, 0, stream>>>(hb, w0bT, fc0_b, h, Nn, D2DIM, D2DIM,
                                                 D2DIM, D2DIM, D2DIM, D2DIM);
  }
  k_bn_relu<<<cgrid((long)Nn * D2DIM, 256), 256, 0, stream>>>(h, hb, bng, bnb, bnm, bnv,
                                                              (long)Nn * D2DIM);

  // ---- degrees (structure-only, once) ----
  k_deg_init<<<cgrid(Nn, 256), 256, 0, stream>>>(deg, Nn);
  k_deg_acc<<<cgrid(E, 256), 256, 0, stream>>>(edges + E, deg, E);
  k_dinv<<<cgrid(Nn, 256), 256, 0, stream>>>(deg, dinv, Nn);

  k_cast_bf16_t<<<cgrid((long)D2DIM * OUTD, 256), 256, 0, stream>>>(fow, fowbT, D2DIM, OUTD);

  for (int layer = 0; layer < 2; ++layer) {
    const size_t wo = (size_t)layer * D2DIM * QKVD;
    k_cast_bf16_t<<<cgrid((long)D2DIM * QKVD, 256), 256, 0, stream>>>(
        wq + wo, wqkvT, D2DIM, QKVD);
    k_cast_bf16_t<<<cgrid((long)D2DIM * QKVD, 256), 256, 0, stream>>>(
        wk + wo, wqkvT + (size_t)QKVD * D2DIM, D2DIM, QKVD);
    k_cast_bf16_t<<<cgrid((long)D2DIM * QKVD, 256), 256, 0, stream>>>(
        wv + wo, wqkvT + (size_t)2 * QKVD * D2DIM, D2DIM, QKVD);
    k_cast_bf16_t<<<cgrid((long)D2DIM * HCDIM, 256), 256, 0, stream>>>(
        gcn_w + (size_t)layer * D2DIM * HCDIM, gwbT, D2DIM, HCDIM);
    k_pack3<<<cgrid(QKV3, 256), 256, 0, stream>>>(bq + layer * QKVD, bk + layer * QKVD,
                                                  bv + layer * QKVD, biasp, QKVD);

    // fused q|k|v projection
    {
      dim3 g((Nn + BM - 1) / BM, QKV3 / BN, 1);
      gemm_wmma_bf16<false><<<g, 128, 0, stream>>>(hb, wqkvT, biasp, qkv, Nn, QKV3,
                                                   D2DIM, D2DIM, D2DIM, QKV3, D2DIM);
    }

    (void)hipMemsetAsync(ksum, 0, (size_t)QKVD * 4, stream);
    (void)hipMemsetAsync(vsum, 0, (size_t)QKVD * 4, stream);
    k_norm_qkv<<<cgrid((long)Nn * NHEAD * 32, 128), 128, 0, stream>>>(qkv, qnb, knT, vT,
                                                                      ksum, vsum, Nn);

    // PT[h] = kvs[h]^T = v^T k (split-K atomic over nodes)
    (void)hipMemsetAsync(PT, 0, (size_t)NHEAD * HCDIM * HCDIM * 4, stream);
    for (int hh = 0; hh < NHEAD; ++hh) {
      dim3 g(HCDIM / BM, HCDIM / BN, (unsigned)((Nn + KCH - 1) / KCH));
      gemm_wmma_bf16<true><<<g, 128, 0, stream>>>(
          vT + (size_t)hh * HCDIM * Nn, knT + (size_t)hh * HCDIM * Nn, nullptr,
          PT + (size_t)hh * HCDIM * HCDIM, HCDIM, HCDIM, Nn, Nn, Nn, HCDIM, KCH);
    }
    k_cast_bf16<<<cgrid((long)NHEAD * HCDIM * HCDIM, 256), 256, 0, stream>>>(
        PT, PTb, (long)NHEAD * HCDIM * HCDIM);

    // num[:,h,:] = q_n @ kvs[h]
    for (int hh = 0; hh < NHEAD; ++hh) {
      dim3 g((Nn + BM - 1) / BM, HCDIM / BN, 1);
      gemm_wmma_bf16<false><<<g, 128, 0, stream>>>(
          qnb + hh * HCDIM, PTb + (size_t)hh * HCDIM * HCDIM, nullptr, num + hh * HCDIM,
          Nn, HCDIM, HCDIM, QKVD, HCDIM, QKVD, HCDIM);
    }
    k_den<<<cgrid((long)Nn * NHEAD * 32, 128), 128, 0, stream>>>(qnb, ksum, den, Nn);
    k_combine<<<cgrid((long)Nn * HCDIM, 256), 256, 0, stream>>>(num, vsum, den, x1, Nn);

    // GCN branch
    {
      dim3 g((Nn + BM - 1) / BM, HCDIM / BN, 1);
      gemm_wmma_bf16<false><<<g, 128, 0, stream>>>(hb, gwbT, nullptr, xw, Nn, HCDIM,
                                                   D2DIM, D2DIM, D2DIM, HCDIM, D2DIM);
    }
    k_gcn_self<<<cgrid((long)Nn * HCDIM, 256), 256, 0, stream>>>(xw, dinv, gout, Nn);
    k_gcn_scatter<<<cgrid((long)E * 32, 256), 256, 0, stream>>>(edges, edges + E, xw,
                                                                dinv, gout, E);

    // concat + residual + bn + relu
    k_concat_bn_relu<<<cgrid((long)Nn * D2DIM, 256), 256, 0, stream>>>(
        h, hb, x1, gout, gcn_b + layer * HCDIM, bng + (layer + 1) * D2DIM,
        bnb + (layer + 1) * D2DIM, bnm + (layer + 1) * D2DIM, bnv + (layer + 1) * D2DIM,
        (long)Nn * D2DIM);
  }

  // ---- output projection ----
  {
    dim3 g((Nn + BM - 1) / BM, (OUTD + BN - 1) / BN, 1);
    gemm_wmma_bf16<false><<<g, 128, 0, stream>>>(hb, fowbT, fobv, out, Nn, OUTD, D2DIM,
                                                 D2DIM, D2DIM, OUTD, D2DIM);
  }
}
